// PBATransformerSparseMLPSession_44049184588218
// MI455X (gfx1250) — compile-verified, matrix operand order checked by reading, not run on hardware
//
#include <hip/hip_runtime.h>

#define D_MODEL 1024
#define D_FF    4096
#define NE      8
#define DIN     1152
#define NTOK    16384
#define S_LEN   2048
#define SESS_CLAMP 51   // NUM_SESSION + 1

typedef __bf16 bf16;
typedef __attribute__((ext_vector_type(16))) __bf16 v16bf;
typedef __attribute__((ext_vector_type(8)))  float  v8f;
typedef __attribute__((ext_vector_type(4)))  int    v4i;

union Frag { uint4 q[2]; v16bf v; };

// ---- workspace layout (bytes) ----
#define OFF_SMAX   0u
#define OFF_CNT    64u
#define OFF_BUCKET 1024u
#define OFF_H      (1024u + (unsigned)NE * NTOK * 4u)            // 525312
#define OFF_FF     (OFF_H + (unsigned)NTOK * DIN * 2u)           // 38274048

// ---- CDNA5 async global->LDS path (guarded: falls back to reg staging) ----
#if defined(__has_builtin)
#if __has_builtin(__builtin_amdgcn_global_load_async_to_lds_b128) && \
    __has_builtin(__builtin_amdgcn_s_wait_asynccnt)
#define USE_ASYNC_LDS 1
#endif
#endif
#ifndef USE_ASYNC_LDS
#define USE_ASYNC_LDS 0
#endif

#if USE_ASYNC_LDS
// param0: global (AS1) pointer to int4; param1: LDS (AS3) pointer to int4
#define GPTR(p) ((__attribute__((address_space(1))) v4i*)(p))
#define LPTR(p) ((__attribute__((address_space(3))) v4i*)(p))
#define ASYNC_B128(gsrc, ldst) \
  __builtin_amdgcn_global_load_async_to_lds_b128(GPTR(gsrc), LPTR(ldst), 0, 0)
#endif

__device__ __forceinline__ int imax(int a, int b) { return a > b ? a : b; }
__device__ __forceinline__ int imin(int a, int b) { return a < b ? a : b; }

__device__ __forceinline__ unsigned short f2bf(float f) {
  union { float f; unsigned u; } v; v.f = f;
  unsigned u = v.u;
  unsigned r = ((u >> 16) & 1u) + 0x7FFFu;   // round-to-nearest-even
  return (unsigned short)((u + r) >> 16);
}
__device__ __forceinline__ unsigned pack2(float a, float b) {
  return (unsigned)f2bf(a) | ((unsigned)f2bf(b) << 16);
}
__device__ __forceinline__ v16bf ldfrag(const bf16* p) {
  Frag f;
  f.q[0] = *(const uint4*)(p);
  f.q[1] = *(const uint4*)(p + 16);
  return f.v;
}

// ---------------- kernel 1: per-row session max + zero counters --------------
__global__ void k_rowmax(const int* __restrict__ sidx, int* __restrict__ smax,
                         int* __restrict__ cnt) {
  __shared__ int red[256];
  int b = blockIdx.x;
  const int* row = sidx + (size_t)b * S_LEN;
  int m = -2147483647 - 1;
  for (int i = threadIdx.x; i < S_LEN; i += 256) m = imax(m, row[i]);
  red[threadIdx.x] = m;
  __syncthreads();
  for (int s = 128; s > 0; s >>= 1) {
    if ((int)threadIdx.x < s) red[threadIdx.x] = imax(red[threadIdx.x], red[threadIdx.x + s]);
    __syncthreads();
  }
  if (threadIdx.x == 0) smax[b] = red[0];
  if (b == 0 && threadIdx.x < NE) cnt[threadIdx.x] = 0;
}

// ---------------- kernel 2: routing (bucket append per expert) ---------------
__global__ void k_route(const int* __restrict__ pos, int* __restrict__ cnt,
                        int* __restrict__ bucket) {
  int t = blockIdx.x * blockDim.x + threadIdx.x;
  if (t >= NTOK) return;
  int e = pos[t];
  int slot = atomicAdd(&cnt[e], 1);
  bucket[e * NTOK + slot] = t;
}

// ---------------- kernel 3: build concat input in bf16 -----------------------
__global__ void k_build(const float* __restrict__ hid,
                        const float* __restrict__ beh_emb,
                        const float* __restrict__ sess_emb,
                        const int* __restrict__ bidx, const int* __restrict__ sidx,
                        const int* __restrict__ smax, unsigned* __restrict__ hout) {
  int t = blockIdx.x;
  int b = t >> 11;                     // S = 2048
  int bi = bidx[t];
  int si = sidx[t];
  int s = -si + smax[b] + 2;
  if (si == -2) s = 0;
  if (si == -1) s = 1;
  if (s > SESS_CLAMP) s = SESS_CLAMP;
  if (s < 0) s = 0;
  const float* src0 = hid + (size_t)t * D_MODEL;
  const float* srcB = beh_emb + bi * 64;
  const float* srcS = sess_emb + s * 64;
  unsigned* dst = hout + (size_t)t * (DIN / 2);
  for (int j = threadIdx.x; j < DIN / 2; j += blockDim.x) {
    int e0 = 2 * j;
    float x0, x1;
    if (e0 < D_MODEL)           { x0 = src0[e0];              x1 = src0[e0 + 1]; }
    else if (e0 < D_MODEL + 64) { x0 = srcB[e0 - D_MODEL];    x1 = srcB[e0 - D_MODEL + 1]; }
    else                        { x0 = srcS[e0 - D_MODEL - 64]; x1 = srcS[e0 - D_MODEL - 63]; }
    dst[j] = pack2(x0, x1);
  }
}

// ---------------- tiled bf16 WMMA GEMM, double-buffered pipeline -------------
// Tile: M=128 (8 waves x 16 rows), N=128 (8x 16-wide accumulators/wave), K=32.
// Clamped token gather (no divergence); weight tile f32->bf16 transposed into
// LDS so both fragments use the ISA 16-bit A layout (lane=M/N&15, kb=(lane>>4)*8).
#define RS 48    // padded LDS row stride in bf16 elems (96 B, 16 B aligned)
#define MT 128
#define NT 128
#define KS 32

template <int KTOT, int NTOT, bool OUT_BF16>
__global__ __launch_bounds__(256) void k_gemm(
    const bf16* __restrict__ Aall, int astride,
    const float* __restrict__ Wall,
    const int* __restrict__ cnt, const int* __restrict__ bucket,
    bf16* __restrict__ outb, float* __restrict__ outf, int ostride) {
  constexpr int NSTEP = KTOT / KS;
  __shared__ bf16 As[2][MT * RS];
  __shared__ bf16 Bs[2][NT * RS];

  const int e = blockIdx.y >> 7;
  const int mTile = blockIdx.y & 127;
  const int count = cnt[e];
  const int mBase = mTile * MT;
  if (mBase >= count) return;
  const int n0 = blockIdx.x * NT;

  const int* buck = bucket + e * NTOK;
  const float* W = Wall + (size_t)e * KTOT * NTOT;

  const int tid = threadIdx.x;
  const int lane = tid & 31;
  const int wv = tid >> 5;
  const int lrow = lane & 15;
  const int kb = (lane >> 4) * 8;      // fragment K base (ISA 16-bit A layout)

  // ---- constant per-thread load geometry (hoisted out of the K loop) ----
  // A: rows tid>>2 and +64, 16B chunk part=tid&3; clamped gather (no branch)
  const int arow = tid >> 2, apart = tid & 3;
  const int tok0 = buck[imin(mBase + arow, count - 1)];
  const int tok1 = buck[imin(mBase + arow + 64, count - 1)];
  const bf16* __restrict__ ap0 = Aall + (size_t)tok0 * astride + apart * 8;
  const bf16* __restrict__ ap1 = Aall + (size_t)tok1 * astride + apart * 8;
  const int aoff0 = arow * RS + apart * 8;
  const int aoff1 = aoff0 + 64 * RS;
  // B: K-row pair kp,kp+1 (kp=(tid&15)*2), 8 cols at nn=(tid>>4)*8
  const int kp = (tid & 15) * 2;
  const int nn = (tid >> 4) * 8;
  const float* __restrict__ wq = W + (size_t)kp * NTOT + n0 + nn;

  v8f acc[8] = {};

  // ---- prologue: stage K-step 0 into buffer 0 ----
  {
#if USE_ASYNC_LDS
    ASYNC_B128(ap0, &As[0][aoff0]);
    ASYNC_B128(ap1, &As[0][aoff1]);
#else
    uint4 ra0 = *(const uint4*)ap0;
    uint4 ra1 = *(const uint4*)ap1;
#endif
    float4 w0 = *(const float4*)(wq);
    float4 w1 = *(const float4*)(wq + 4);
    float4 w2 = *(const float4*)(wq + NTOT);
    float4 w3 = *(const float4*)(wq + NTOT + 4);
#if !USE_ASYNC_LDS
    *(uint4*)&As[0][aoff0] = ra0;
    *(uint4*)&As[0][aoff1] = ra1;
#endif
    bf16* bb = &Bs[0][0];
    *(unsigned*)&bb[(nn + 0) * RS + kp] = pack2(w0.x, w2.x);
    *(unsigned*)&bb[(nn + 1) * RS + kp] = pack2(w0.y, w2.y);
    *(unsigned*)&bb[(nn + 2) * RS + kp] = pack2(w0.z, w2.z);
    *(unsigned*)&bb[(nn + 3) * RS + kp] = pack2(w0.w, w2.w);
    *(unsigned*)&bb[(nn + 4) * RS + kp] = pack2(w1.x, w3.x);
    *(unsigned*)&bb[(nn + 5) * RS + kp] = pack2(w1.y, w3.y);
    *(unsigned*)&bb[(nn + 6) * RS + kp] = pack2(w1.z, w3.z);
    *(unsigned*)&bb[(nn + 7) * RS + kp] = pack2(w1.w, w3.w);
  }
#if USE_ASYNC_LDS
  __builtin_amdgcn_s_wait_asynccnt(0);   // own async writes done before barrier
#endif
  __syncthreads();

  for (int ks = 0; ks < NSTEP; ++ks) {
    const int cur = ks & 1, nxt = cur ^ 1;
    const bool more = (ks + 1 < NSTEP);
    const int k1 = (ks + 1) * KS;

    // ---- issue next tile's global loads before the math ----
    uint4 ra0, ra1;
    float4 w0, w1, w2, w3;
    if (more) {
#if USE_ASYNC_LDS
      ASYNC_B128(ap0 + k1, &As[nxt][aoff0]);
      ASYNC_B128(ap1 + k1, &As[nxt][aoff1]);
#else
      ra0 = *(const uint4*)(ap0 + k1);
      ra1 = *(const uint4*)(ap1 + k1);
#endif
      wq += (size_t)KS * NTOT;
      w0 = *(const float4*)(wq);
      w1 = *(const float4*)(wq + 4);
      w2 = *(const float4*)(wq + NTOT);
      w3 = *(const float4*)(wq + NTOT + 4);
      if (ks + 2 < NSTEP)   // stream the slab after next toward L2
        __builtin_prefetch((const void*)(wq + (size_t)KS * NTOT), 0, 0);
    }

    // ---- compute on buffer cur: 1 A fragment x 8 B fragments ----
    v16bf afrag = ldfrag(&As[cur][(wv * 16 + lrow) * RS + kb]);
#pragma unroll
    for (int j = 0; j < 8; ++j) {
      v16bf bfrag = ldfrag(&Bs[cur][(j * 16 + lrow) * RS + kb]);
      acc[j] = __builtin_amdgcn_wmma_f32_16x16x32_bf16(
          false, afrag, false, bfrag, (short)0, acc[j], false, false);
    }

    // ---- drain staged registers into buffer nxt ----
    if (more) {
#if !USE_ASYNC_LDS
      *(uint4*)&As[nxt][aoff0] = ra0;
      *(uint4*)&As[nxt][aoff1] = ra1;
#endif
      bf16* bb = &Bs[nxt][0];
      *(unsigned*)&bb[(nn + 0) * RS + kp] = pack2(w0.x, w2.x);
      *(unsigned*)&bb[(nn + 1) * RS + kp] = pack2(w0.y, w2.y);
      *(unsigned*)&bb[(nn + 2) * RS + kp] = pack2(w0.z, w2.z);
      *(unsigned*)&bb[(nn + 3) * RS + kp] = pack2(w0.w, w2.w);
      *(unsigned*)&bb[(nn + 4) * RS + kp] = pack2(w1.x, w3.x);
      *(unsigned*)&bb[(nn + 5) * RS + kp] = pack2(w1.y, w3.y);
      *(unsigned*)&bb[(nn + 6) * RS + kp] = pack2(w1.z, w3.z);
      *(unsigned*)&bb[(nn + 7) * RS + kp] = pack2(w1.w, w3.w);
    }
#if USE_ASYNC_LDS
    __builtin_amdgcn_s_wait_asynccnt(0);
#endif
    __syncthreads();
  }

  // ---- epilogue: C layout — lane holds N=lane&15, rows M=(lane>>4)*8+i ----
  const int mrow = wv * 16 + (lane >> 4) * 8;
  int toks[8];
  bool vld[8];
#pragma unroll
  for (int i = 0; i < 8; ++i) {
    int m = mBase + mrow + i;
    vld[i] = (m < count);
    toks[i] = buck[imin(m, count - 1)];
  }
#pragma unroll
  for (int j = 0; j < 8; ++j) {
    int col = n0 + j * 16 + lrow;
#pragma unroll
    for (int i = 0; i < 8; ++i) {
      if (!vld[i]) continue;
      float v = acc[j][i];
      if constexpr (OUT_BF16) {
        v = fmaxf(v, 0.0f);   // fused relu
        ((unsigned short*)outb)[(size_t)toks[i] * ostride + col] = f2bf(v);
      } else {
        outf[(size_t)toks[i] * ostride + col] = v;
      }
    }
  }
}

extern "C" void kernel_launch(void* const* d_in, const int* in_sizes, int n_in,
                              void* d_out, int out_size, void* d_ws, size_t ws_size,
                              hipStream_t stream) {
  const float* hidden   = (const float*)d_in[0];
  const float* wi       = (const float*)d_in[1];
  const float* wo       = (const float*)d_in[2];
  const float* beh_emb  = (const float*)d_in[3];
  const float* sess_emb = (const float*)d_in[4];
  const int*   pos      = (const int*)d_in[5];
  const int*   bidx     = (const int*)d_in[6];
  const int*   sidx     = (const int*)d_in[7];
  float* out = (float*)d_out;

  char* ws = (char*)d_ws;
  int*      smax   = (int*)(ws + OFF_SMAX);
  int*      cnt    = (int*)(ws + OFF_CNT);
  int*      bucket = (int*)(ws + OFF_BUCKET);
  unsigned* hbuf   = (unsigned*)(ws + OFF_H);
  bf16*     hbf    = (bf16*)(ws + OFF_H);
  bf16*     ffbf   = (bf16*)(ws + OFF_FF);

  k_rowmax<<<8, 256, 0, stream>>>(sidx, smax, cnt);
  k_route<<<NTOK / 256, 256, 0, stream>>>(pos, cnt, bucket);
  k_build<<<NTOK, 128, 0, stream>>>(hidden, beh_emb, sess_emb, bidx, sidx, smax, hbuf);
  // GEMM1: relu(h @ wi[e]) -> ff (bf16). N tiles = 4096/128, M = 8 experts x 128 tiles
  k_gemm<DIN, D_FF, true><<<dim3(D_FF / NT, NE * 128), 256, 0, stream>>>(
      hbf, DIN, wi, cnt, bucket, ffbf, nullptr, D_FF);
  // GEMM2: ff @ wo[e] -> out (f32 scatter by token)
  k_gemm<D_FF, D_MODEL, false><<<dim3(D_MODEL / NT, NE * 128), 256, 0, stream>>>(
      ffbf, D_FF, wo, cnt, bucket, nullptr, out, D_MODEL);
}
